// tworing_conv_layer_batch_50543175139553
// MI455X (gfx1250) — matrix operand order repaired and checked by reading, not compiled, not run on hardware
//
#include <hip/hip_runtime.h>

typedef _Float16 f16;
typedef _Float16 v16h __attribute__((ext_vector_type(16)));
typedef _Float16 v8h  __attribute__((ext_vector_type(8)));
typedef float    v8f  __attribute__((ext_vector_type(8)));
typedef int      v4i  __attribute__((ext_vector_type(4)));

#define NV    40962
#define KR    19
#define CIN   64
#define COUT  64
#define BATCH 4
#define KTOT  (KR * CIN)    // 1216
#define NKB   (KTOT / 32)   // 38 K-blocks of 32
#define TILE_N 128

// ---- CDNA5 async global->LDS path (guarded; fallback = load + ds_store) ----
#if defined(__has_builtin)
#  if __has_builtin(__builtin_amdgcn_global_load_async_to_lds_b128)
#    define HAVE_ASYNC_LDS 1
#  endif
#endif
#ifndef HAVE_ASYNC_LDS
#  define HAVE_ASYNC_LDS 0
#endif

#if HAVE_ASYNC_LDS
typedef __attribute__((address_space(1))) v4i* gv4i_p;  // global int4*
typedef __attribute__((address_space(3))) v4i* lv4i_p;  // LDS int4*
__device__ __forceinline__ gv4i_p to_global(const void* p) {
  return (gv4i_p)(unsigned long long)p;
}
__device__ __forceinline__ lv4i_p to_lds(void* p) {
  // generic pointers to LDS carry the LDS byte offset in their low 32 bits
  return (lv4i_p)(unsigned int)(unsigned long long)p;
}
__device__ __forceinline__ void wait_async_all() {
#  if __has_builtin(__builtin_amdgcn_s_wait_asynccnt)
  __builtin_amdgcn_s_wait_asynccnt(0);
#  else
  asm volatile("s_wait_asynccnt 0x0" ::: "memory");
#  endif
}
#else
__device__ __forceinline__ void wait_async_all() {}
#endif

// ---------------------------------------------------------------------------
// prep_x: x (B, C, N) f32  ->  xh/xl (B, N, C) f16 hi/lo split
// ---------------------------------------------------------------------------
__global__ void prep_x(const float* __restrict__ x,
                       f16* __restrict__ xh, f16* __restrict__ xl) {
  int t = blockIdx.x * blockDim.x + threadIdx.x;
  if (t >= BATCH * NV) return;
  int b = t / NV, n = t % NV;
  f16 hb[CIN], lb[CIN];
#pragma unroll
  for (int c = 0; c < CIN; c++) {
    float v = x[((size_t)b * CIN + c) * NV + n];
    f16 h = (f16)v;
    f16 l = (f16)(v - (float)h);
    hb[c] = h; lb[c] = l;
  }
  f16* ph = xh + ((size_t)b * NV + n) * CIN;
  f16* pl = xl + ((size_t)b * NV + n) * CIN;
#pragma unroll
  for (int i = 0; i < 8; i++) {
    *(v8h*)(ph + i * 8) = *(v8h*)(hb + i * 8);
    *(v8h*)(pl + i * 8) = *(v8h*)(lb + i * 8);
  }
}

// ---------------------------------------------------------------------------
// prep_w: W (64 x 1216) f32 -> hi/lo f16, pre-swizzled into WMMA A-operand
// lane layout (contiguous k = 16*(lane/16)+e, not the ISA table order). Fragment t = (kb*4+m)*32+lane,
// 32B contiguous per lane.
// ---------------------------------------------------------------------------
__global__ void prep_w(const float* __restrict__ W,
                       f16* __restrict__ wh, f16* __restrict__ wl) {
  int t = blockIdx.x * blockDim.x + threadIdx.x;
  if (t >= NKB * 4 * 32) return;
  int lane = t & 31;
  int m    = (t >> 5) & 3;
  int kb   = t >> 7;
  int row   = m * 16 + (lane & 15);
  int kbase = kb * 32 + ((lane < 16) ? 0 : 16);
  f16 hb[16], lb[16];
#pragma unroll
  for (int e = 0; e < 16; e++) {
    int kk = kbase + e;
    float v = W[(size_t)row * KTOT + kk];
    f16 h = (f16)v;
    f16 l = (f16)(v - (float)h);
    hb[e] = h; lb[e] = l;
  }
  *(v16h*)(wh + (size_t)t * 16) = *(v16h*)hb;
  *(v16h*)(wl + (size_t)t * 16) = *(v16h*)lb;
}

// ---------------------------------------------------------------------------
// conv_main: block = (batch, 128-vertex tile), all 64 couts. 8 waves:
// m = wid&3 (16-cout strip), nh = wid>>2 (64-vertex half). Double-buffered
// LDS staging of the gathered 128x64 hi/lo chunk; on CDNA5 the gather goes
// straight to LDS via global_load_async_to_lds_b128 (ASYNCcnt), fully
// overlapped with the WMMAs on the other buffer.
// ---------------------------------------------------------------------------
__global__ __launch_bounds__(256) void conv_main(
    const f16* __restrict__ xh, const f16* __restrict__ xl,
    const f16* __restrict__ wh, const f16* __restrict__ wl,
    const int* __restrict__ neigh, const float* __restrict__ bias,
    float* __restrict__ out) {
  __shared__ __align__(128) f16 ldsBh[2 * TILE_N * CIN];  // 2 x 16 KB
  __shared__ __align__(128) f16 ldsBl[2 * TILE_N * CIN];  // 2 x 16 KB

  const int tid  = threadIdx.x;
  const int lane = tid & 31;
  const int wid  = tid >> 5;
  const int m    = wid & 3;
  const int nh   = wid >> 2;
  const int b    = blockIdx.y;
  const int n0   = blockIdx.x * TILE_N;

  // staging role: thread pair per vertex (half = lower/upper 32 channels)
  const int vload = tid >> 1;
  const int half  = tid & 1;
  int vglob = n0 + vload;
  if (vglob > NV - 1) vglob = NV - 1;  // clamp: keeps EXEC all-ones for WMMA
  const int* idxrow = neigh + (size_t)vglob * KR;

  const int col    = lane & 15;
  const int hiHalf = (lane >= 16) ? 1 : 0;
  const int loff   = vload * CIN + half * 32;  // f16 units within a buffer

  v8f acc[4] = {};

  // stage neighbor k's gathered chunk (64B hi + 64B lo per thread) into buf
  auto stage = [&](int k, int buf) {
    int nb = idxrow[k];
    const f16* gph = xh + (((size_t)b * NV + nb) * CIN + half * 32);
    const f16* gpl = xl + (((size_t)b * NV + nb) * CIN + half * 32);
    f16* dph = ldsBh + buf * (TILE_N * CIN) + loff;
    f16* dpl = ldsBl + buf * (TILE_N * CIN) + loff;
#if HAVE_ASYNC_LDS
    gv4i_p gH = to_global(gph);
    gv4i_p gL = to_global(gpl);
    lv4i_p lH = to_lds(dph);
    lv4i_p lL = to_lds(dpl);
    // ISA: IOFFSET applies to BOTH global and LDS addresses for async loads
    __builtin_amdgcn_global_load_async_to_lds_b128(gH, lH, 0, 0);
    __builtin_amdgcn_global_load_async_to_lds_b128(gH, lH, 16, 0);
    __builtin_amdgcn_global_load_async_to_lds_b128(gH, lH, 32, 0);
    __builtin_amdgcn_global_load_async_to_lds_b128(gH, lH, 48, 0);
    __builtin_amdgcn_global_load_async_to_lds_b128(gL, lL, 0, 0);
    __builtin_amdgcn_global_load_async_to_lds_b128(gL, lL, 16, 0);
    __builtin_amdgcn_global_load_async_to_lds_b128(gL, lL, 32, 0);
    __builtin_amdgcn_global_load_async_to_lds_b128(gL, lL, 48, 0);
#else
    v16h gh0 = *(const v16h*)(gph);
    v16h gh1 = *(const v16h*)(gph + 16);
    v16h gl0 = *(const v16h*)(gpl);
    v16h gl1 = *(const v16h*)(gpl + 16);
    *(v16h*)(dph)      = gh0;
    *(v16h*)(dph + 16) = gh1;
    *(v16h*)(dpl)      = gl0;
    *(v16h*)(dpl + 16) = gl1;
#endif
  };

  stage(0, 0);  // prologue

  for (int k = 0; k < KR; k++) {
    const int cur = k & 1;
    wait_async_all();   // this wave's async LDS writes (buffer `cur`) done
    __syncthreads();    // all waves: staging visible, prev compute finished

    if (k + 1 < KR) stage(k + 1, cur ^ 1);  // overlaps with compute below

    const f16* bhB = ldsBh + cur * (TILE_N * CIN);
    const f16* blB = ldsBl + cur * (TILE_N * CIN);
#pragma unroll
    for (int ks = 0; ks < 2; ks++) {
      const int kb = 2 * k + ks;
      const f16* aph = wh + ((size_t)(kb * 4 + m) * 32 + lane) * 16;
      const f16* apl = wl + ((size_t)(kb * 4 + m) * 32 + lane) * 16;
      v16h ah = *(const v16h*)aph;
      v16h al = *(const v16h*)apl;
#pragma unroll
      for (int st = 0; st < 4; st++) {
        // B operand (32x16 KxN): lane = column n, 16 consecutive channels
        const int boff = (nh * 64 + st * 16 + col) * CIN + ks * 32 + hiHalf * 16;
        v16h bh = *(const v16h*)(bhB + boff);
        v16h bl = *(const v16h*)(blB + boff);
        acc[st] = __builtin_amdgcn_wmma_f32_16x16x32_f16(
            false, ah, false, bh, (short)0, acc[st], false, false);
        acc[st] = __builtin_amdgcn_wmma_f32_16x16x32_f16(
            false, ah, false, bl, (short)0, acc[st], false, false);
        acc[st] = __builtin_amdgcn_wmma_f32_16x16x32_f16(
            false, al, false, bh, (short)0, acc[st], false, false);
      }
    }
  }

  // ---- bias + store (C/D layout: VGPR j, lane L -> row j + (L<16?0:8)) ----
  const int rbase = m * 16 + (hiHalf ? 8 : 0);
  float bv[8];
#pragma unroll
  for (int j = 0; j < 8; j++) bv[j] = bias[rbase + j];
#pragma unroll
  for (int st = 0; st < 4; st++) {
    int nvert = n0 + nh * 64 + st * 16 + col;
    if (nvert < NV) {
#pragma unroll
      for (int j = 0; j < 8; j++) {
        out[((size_t)b * COUT + (rbase + j)) * NV + nvert] = acc[st][j] + bv[j];
      }
    }
  }
}

// ---------------------------------------------------------------------------
extern "C" void kernel_launch(void* const* d_in, const int* in_sizes, int n_in,
                              void* d_out, int out_size, void* d_ws, size_t ws_size,
                              hipStream_t stream) {
  const float* x     = (const float*)d_in[0];  // (4, 64, 40962) f32
  const int*   neigh = (const int*)d_in[1];    // (40962, 19) i32
  const float* W     = (const float*)d_in[2];  // (64, 1216) f32
  const float* bias  = (const float*)d_in[3];  // (64,) f32
  float*       out   = (float*)d_out;          // (4, 64, 40962) f32

  const size_t XSZ = (size_t)BATCH * NV * CIN;  // f16 elements per x array
  const size_t WSZ = (size_t)NKB * 4 * 32 * 16; // f16 elements per W array
  f16* xh = (f16*)d_ws;
  f16* xl = xh + XSZ;
  f16* wh = xl + XSZ;
  f16* wl = wh + WSZ;   // total ws use: 2*XSZ*2 + 2*WSZ*2 bytes (~42.3 MB)

  hipLaunchKernelGGL(prep_x, dim3((BATCH * NV + 255) / 256), dim3(256), 0, stream,
                     x, xh, xl);
  hipLaunchKernelGGL(prep_w, dim3((NKB * 4 * 32 + 255) / 256), dim3(256), 0, stream,
                     W, wh, wl);

  dim3 grid((NV + TILE_N - 1) / TILE_N, BATCH);
  hipLaunchKernelGGL(conv_main, grid, dim3(256), 0, stream,
                     xh, xl, wh, wl, neigh, bias, out);
}
